// Net_67774583931399
// MI455X (gfx1250) — matrix operand order repaired and checked by reading, not compiled, not run on hardware
//
#include <hip/hip_runtime.h>
#include <hip/hip_bf16.h>

// SNN: 26-step LIF MLP 784->200->50->10, B=32768.
// Fully fused persistent-tile kernel: each block owns 64 batch rows.
// Phase 1: cur1 = f16-WMMA GEMM of x[64x784] @ W1^T (dropconnect applied) + b1 -> LDS.
// Phase 2: 26 steps, all state in LDS; layer2 via f16 WMMA on binary spikes; layer3 scalar.
// HBM traffic: x read once (102.8MB) + outputs (68MB) => ~7.3us floor @ 23.3TB/s.

typedef __attribute__((ext_vector_type(16))) _Float16 v16h;
typedef __attribute__((ext_vector_type(8)))  float    v8f;

union FragU { v16h v; unsigned int u[8]; };

#define MT        64      // batch rows per block
#define H1        200
#define H1P       224     // padded K for layer-2 WMMA (7 chunks of 32)
#define H2        50
#define H2P       64      // padded N for layer-2 WMMA (4 tiles of 16)
#define NOUT      10
#define NIN       784
#define NSTEPS    26
#define BETA      0.95f
#define THR       1.0f

struct SnnLds {
  float    cur1[MT * H1];     // 51,200 B
  float    mem1[MT * H1];     // 51,200 B
  _Float16 spk1[MT * H1P];    // 28,672 B (cols >=200 zero)
  _Float16 w2  [H2P * H1P];   // 28,672 B  n-major: w2[n][k] = W2[n][k], padded 0
  float    mem2[MT * H2];     // 12,800 B
  float    spk2[MT * H2];     // 12,800 B
  float    mem3[MT * NOUT];   //  2,560 B
  float    w3  [NOUT * H2];   //  2,000 B
  float    b1s [H1];          //    800 B
  float    b2s [H2P];         //    256 B (pad zero)
  float    b3s [16];          //     64 B
  _Float16 xs  [MT * 32];     //  4,096 B  phase-1 A stage (one K chunk)
  _Float16 ws1 [208 * 32];    // 13,312 B  phase-1 B stage (n-major, one K chunk)
};                            // total ~208 KB of the 320 KB WGP LDS

__device__ __forceinline__ v8f wmma_f16(const FragU& a, const FragU& b, v8f c) {
  return __builtin_amdgcn_wmma_f32_16x16x32_f16(false, a.v, false, b.v,
                                                (short)0, c, false, false);
}

extern "C" __global__ void __launch_bounds__(256, 1)
snn_fused_kernel(const float* __restrict__ x,
                 const float* __restrict__ W1, const float* __restrict__ b1,
                 const float* __restrict__ W2, const float* __restrict__ b2,
                 const float* __restrict__ W3, const float* __restrict__ b3,
                 float* __restrict__ out, int batch) {
  extern __shared__ char smem_raw[];
  SnnLds& L = *reinterpret_cast<SnnLds*>(smem_raw);

  const int tid   = threadIdx.x;
  const int wave  = tid >> 5;          // wave32
  const int lane  = tid & 31;
  const int lhalf = lane >> 4;         // 0: lanes 0-15, 1: lanes 16-31
  const int l16   = lane & 15;
  const int m0    = blockIdx.x * MT;
  const size_t rec = (size_t)NSTEPS * (size_t)batch * NOUT; // spk block size in out

  // ---------------- init LDS state ----------------
  for (int i = tid; i < H2P * H1P; i += 256) {
    int n = i / H1P, k = i - n * H1P;
    float v = (n < H2 && k < H1) ? W2[n * H1 + k] : 0.f;
    L.w2[i] = (_Float16)v;
  }
  for (int i = tid; i < MT * H1P; i += 256) L.spk1[i] = (_Float16)0.f;
  for (int i = tid; i < MT * H1;  i += 256) L.mem1[i] = 0.f;
  for (int i = tid; i < MT * H2;  i += 256) L.mem2[i] = 0.f;
  for (int i = tid; i < MT * NOUT;i += 256) L.mem3[i] = 0.f;
  for (int i = tid; i < NOUT * H2;i += 256) L.w3[i] = W3[i];
  for (int i = tid; i < H1;       i += 256) L.b1s[i] = b1[i];
  for (int i = tid; i < H2P;      i += 256) L.b2s[i] = (i < H2) ? b2[i] : 0.f;
  for (int i = tid; i < NOUT;     i += 256) L.b3s[i] = b3[i];
  __syncthreads();

  // ---------------- phase 1: cur1 = x @ W1^T + b1 via WMMA ----------------
  // 52 output tiles (4 row-tiles x 13 col-tiles over 208 padded cols), 8 waves.
  v8f acc0 = {}, acc1 = {}, acc2 = {}, acc3 = {}, acc4 = {}, acc5 = {}, acc6 = {};
  const int NCH = (NIN + 31) / 32;  // 25 K-chunks (last partial)

  for (int kc = 0; kc < NCH; ++kc) {
    const int kb = kc * 32;
    // stage x chunk [64 x 32] -> f16
    for (int i = tid; i < MT * 32; i += 256) {
      int m = i >> 5, kk = i & 31, k = kb + kk;
      float v = (k < NIN) ? x[(size_t)(m0 + m) * NIN + k] : 0.f;
      L.xs[i] = (_Float16)v;
    }
    // stage W1 chunk [208 x 32] n-major, with DropConnect threshold
    for (int i = tid; i < 208 * 32; i += 256) {
      int n = i >> 5, kk = i & 31, k = kb + kk;
      float v = 0.f;
      if (n < H1 && k < NIN) {
        v = W1[(size_t)n * NIN + k];
        if (fabsf(v) < 1e-6f) v = 0.f;
      }
      L.ws1[i] = (_Float16)v;
    }
    // prefetch next x chunk into L2 (global_prefetch_b8)
    if (kc + 1 < NCH) {
      int m = tid >> 5, kk = tid & 31, k = kb + 32 + kk;
      if (k < NIN) __builtin_prefetch(&x[(size_t)(m0 + m) * NIN + k], 0, 1);
    }
    __syncthreads();

#pragma unroll
    for (int ti = 0; ti < 7; ++ti) {
      const int t = wave + ti * 8;     // wave-uniform guard (EXEC stays all-1s)
      if (t < 52) {
        const int rt = t & 3, ct = t >> 2;
        FragU a, bfr;
#pragma unroll
        for (int v = 0; v < 8; ++v) {
          const int ka = (v < 4 ? 2 * v : 16 + 2 * (v - 4)) + lhalf * 8;
          a.u[v]   = *(const unsigned int*)&L.xs[(rt * 16 + l16) * 32 + ka];
          const int kbv = (v < 4 ? 2 * v : 16 + 2 * (v - 4)) + lhalf * 8;
          bfr.u[v] = *(const unsigned int*)&L.ws1[(ct * 16 + l16) * 32 + kbv];
        }
        v8f c;
        switch (ti) { case 0: c = acc0; break; case 1: c = acc1; break;
                      case 2: c = acc2; break; case 3: c = acc3; break;
                      case 4: c = acc4; break; case 5: c = acc5; break;
                      default: c = acc6; }
        c = wmma_f16(a, bfr, c);
        switch (ti) { case 0: acc0 = c; break; case 1: acc1 = c; break;
                      case 2: acc2 = c; break; case 3: acc3 = c; break;
                      case 4: acc4 = c; break; case 5: acc5 = c; break;
                      default: acc6 = c; }
      }
    }
    __syncthreads();
  }

  // write cur1 tiles (+ b1) into LDS
#pragma unroll
  for (int ti = 0; ti < 7; ++ti) {
    const int t = wave + ti * 8;
    if (t < 52) {
      const int rt = t & 3, ct = t >> 2;
      const int col = ct * 16 + l16;
      if (col < H1) {
        v8f c;
        switch (ti) { case 0: c = acc0; break; case 1: c = acc1; break;
                      case 2: c = acc2; break; case 3: c = acc3; break;
                      case 4: c = acc4; break; case 5: c = acc5; break;
                      default: c = acc6; }
#pragma unroll
        for (int r = 0; r < 8; ++r) {
          const int row = rt * 16 + lhalf * 8 + r;
          L.cur1[row * H1 + col] = c[r] + L.b1s[col];
        }
      }
    }
  }
  __syncthreads();

  // ---------------- phase 2: 26 LIF steps ----------------
  for (int step = 0; step < NSTEPS; ++step) {
    // layer-1 membrane + spikes (reset uses OLD mem; spike uses NEW mem)
    for (int i = tid; i < MT * H1; i += 256) {
      const int m = i / H1, j = i - m * H1;
      const float old = L.mem1[i];
      const float nm = BETA * old + L.cur1[i] - ((old - THR > 0.f) ? THR : 0.f);
      L.mem1[i] = nm;
      L.spk1[m * H1P + j] = (_Float16)((nm - THR > 0.f) ? 1.0f : 0.0f);
    }
    __syncthreads();

    // layer-2: cur2 = spk1 @ W2^T via WMMA; 16 tiles (4x4), 2 per wave
#pragma unroll
    for (int ti = 0; ti < 2; ++ti) {
      const int t = wave * 2 + ti;
      const int rt = t & 3, ct = t >> 2;
      v8f acc = {};
#pragma unroll
      for (int kc = 0; kc < H1P / 32; ++kc) {   // 7 chunks
        FragU a, bfr;
#pragma unroll
        for (int v = 0; v < 8; ++v) {
          const int ka = kc * 32 + (v < 4 ? 2 * v : 16 + 2 * (v - 4)) + lhalf * 8;
          a.u[v]   = *(const unsigned int*)&L.spk1[(rt * 16 + l16) * H1P + ka];
          const int kbv = kc * 32 + (v < 4 ? 2 * v : 16 + 2 * (v - 4)) + lhalf * 8;
          bfr.u[v] = *(const unsigned int*)&L.w2[(ct * 16 + l16) * H1P + kbv];
        }
        acc = wmma_f16(a, bfr, acc);
      }
      // layer-2 membrane + spikes, directly in WMMA output layout
      const int col = ct * 16 + l16;
      if (col < H2) {
#pragma unroll
        for (int r = 0; r < 8; ++r) {
          const int row = rt * 16 + lhalf * 8 + r;
          const float c2 = acc[r] + L.b2s[col];
          const float old = L.mem2[row * H2 + col];
          const float nm = BETA * old + c2 - ((old - THR > 0.f) ? THR : 0.f);
          L.mem2[row * H2 + col] = nm;
          L.spk2[row * H2 + col] = (nm - THR > 0.f) ? 1.0f : 0.0f;
        }
      }
    }
    __syncthreads();

    // layer-3 (50->10) scalar + record outputs
    for (int i = tid; i < MT * NOUT; i += 256) {
      const int m = i / NOUT, o = i - m * NOUT;
      float s = L.b3s[o];
#pragma unroll 5
      for (int j = 0; j < H2; ++j) s += L.spk2[m * H2 + j] * L.w3[o * H2 + j];
      const float old = L.mem3[i];
      const float nm = BETA * old + s - ((old - THR > 0.f) ? THR : 0.f);
      L.mem3[i] = nm;
      const float spk = (nm - THR > 0.f) ? 1.0f : 0.0f;
      const size_t base = (size_t)step * (size_t)batch * NOUT + (size_t)(m0 + m) * NOUT + o;
      out[base]       = spk;   // spk3_rec
      out[rec + base] = nm;    // mem3_rec
    }
    __syncthreads();
  }
}

extern "C" void kernel_launch(void* const* d_in, const int* in_sizes, int n_in,
                              void* d_out, int out_size, void* d_ws, size_t ws_size,
                              hipStream_t stream) {
  const float* x  = (const float*)d_in[0];
  const float* W1 = (const float*)d_in[1];
  const float* b1 = (const float*)d_in[2];
  const float* W2 = (const float*)d_in[3];
  const float* b2 = (const float*)d_in[4];
  const float* W3 = (const float*)d_in[5];
  const float* b3 = (const float*)d_in[6];
  float* out = (float*)d_out;

  const int batch = in_sizes[0] / NIN;      // 32768
  const int blocks = batch / MT;            // 512
  const size_t lds = sizeof(SnnLds);        // ~208 KB (CDNA5 WGP has 320 KB)

  snn_fused_kernel<<<dim3(blocks), dim3(256), lds, stream>>>(
      x, W1, b1, W2, b2, W3, b3, out, batch);
  (void)d_ws; (void)ws_size; (void)n_in; (void)out_size;
}
